// MultiHeadedAttention_3564822856061
// MI455X (gfx1250) — matrix operand order repaired and checked by reading, not compiled, not run on hardware
//
#include <hip/hip_runtime.h>
#include <stdint.h>

// ---------------------------------------------------------------------------
// MI455X / gfx1250, wave32. bf16 WMMA + Tensor Data Mover staging.
// ---------------------------------------------------------------------------

typedef __attribute__((ext_vector_type(16))) __bf16          v16bf;
typedef __attribute__((ext_vector_type(8)))  float           v8f;
typedef __attribute__((ext_vector_type(8)))  unsigned short  us8;
typedef __attribute__((ext_vector_type(16))) unsigned short  us16;
typedef __attribute__((ext_vector_type(4)))  float           f4;
typedef __attribute__((ext_vector_type(4)))  unsigned int    ui4;
typedef __attribute__((ext_vector_type(8)))  int             i8v;
typedef __attribute__((ext_vector_type(4)))  int             i4v;

#define TT   2048
#define NH   16
#define DKK  64
#define FF   1024
#define BAND 64
#define MINV (-3.402823466e38f)

#if __has_builtin(__builtin_amdgcn_tensor_load_to_lds) && \
    __has_builtin(__builtin_amdgcn_s_wait_tensorcnt)
#define USE_TDM 1
#else
#define USE_TDM 0
#endif

static __device__ __forceinline__ unsigned short f32_to_bf16(float f) {
  union { float f; unsigned int u; } c; c.f = f;
  unsigned int u = c.u;
  return (unsigned short)((u + 0x7FFFu + ((u >> 16) & 1u)) >> 16);  // RNE
}

static __device__ __forceinline__ v16bf make_frag(us8 lo, us8 hi) {
  us16 t = __builtin_shufflevector(lo, hi, 0,1,2,3,4,5,6,7,8,9,10,11,12,13,14,15);
  return __builtin_bit_cast(v16bf, t);
}

static __device__ __forceinline__ v8f v8zero() {
  v8f z;
  #pragma unroll
  for (int i = 0; i < 8; ++i) z[i] = 0.0f;
  return z;
}

#if USE_TDM
static __device__ __forceinline__ unsigned lds_off32(const void* p) {
  return (unsigned)(unsigned long long)(uintptr_t)p;  // LDS aperture: offset in addr[31:0]
}

// TDM 2D tile load: tile_w x tile_h bf16 elements, row stride (elements),
// packed row-major into LDS at lds_addr.  D# per CDNA5 ISA.
// clang-23 builtin form: (uint32x4, int32x8, int32x4, int32x4, int32x8, i32 cpol)
static __device__ __forceinline__ void tdm_load_2d(unsigned lds_addr,
                                                   const unsigned short* gptr,
                                                   unsigned tile_w, unsigned tile_h,
                                                   unsigned row_stride) {
  const unsigned long long ga = (unsigned long long)(uintptr_t)gptr;
  ui4 g0;
  g0[0] = 1u;                                            // count=1, user mode
  g0[1] = lds_addr;                                      // lds_addr [63:32]
  g0[2] = (unsigned)(ga & 0xFFFFFFFFull);                // global_addr [95:64]
  g0[3] = (unsigned)((ga >> 32) & 0x01FFFFFFull)         // global_addr [120:96]
        | (2u << 30);                                    // type=2 ("image")
  i8v g1;
  g1[0] = (int)(1u << 16);                               // data_size=1 (2 bytes)
  g1[1] = (int)((tile_w & 0xFFFFu) << 16);               // tensor_dim0 lo16
  g1[2] = (int)(((tile_w >> 16) & 0xFFFFu)               // tensor_dim0 hi16
        | ((tile_h & 0xFFFFu) << 16));                   // tensor_dim1 lo16
  g1[3] = (int)(((tile_h >> 16) & 0xFFFFu)               // tensor_dim1 hi16
        | ((tile_w & 0xFFFFu) << 16));                   // tile_dim0
  g1[4] = (int)(tile_h & 0xFFFFu);                       // tile_dim1 (tile_dim2=0)
  g1[5] = (int)row_stride;                               // tensor_dim0_stride lo32
  g1[6] = 0;                                             // stride hi / dim1_stride lo
  g1[7] = 0;
  const i4v z4 = {0, 0, 0, 0};
  const i8v z8 = {0, 0, 0, 0, 0, 0, 0, 0};
  __builtin_amdgcn_tensor_load_to_lds(g0, g1, z4, z4, z8, 0);
}
#endif

// ---------------------------------------------------------------------------
// f32 -> bf16 pre-conversion (8 elements/thread, b128 in / b128 out)
// ---------------------------------------------------------------------------
__global__ __launch_bounds__(256)
void cvt_f32_bf16(const float* __restrict__ in, unsigned short* __restrict__ out, int n) {
  const int idx = (blockIdx.x * 256 + threadIdx.x) * 8;
  if (idx >= n) return;
  f4 a = *(const f4*)(in + idx);
  f4 b = *(const f4*)(in + idx + 4);
  us8 o;
  o[0] = f32_to_bf16(a.x); o[1] = f32_to_bf16(a.y);
  o[2] = f32_to_bf16(a.z); o[3] = f32_to_bf16(a.w);
  o[4] = f32_to_bf16(b.x); o[5] = f32_to_bf16(b.y);
  o[6] = f32_to_bf16(b.z); o[7] = f32_to_bf16(b.w);
  *(us8*)(out + idx) = o;
}

// ---------------------------------------------------------------------------
// WMMA tile compute: 2x4 accumulators (32x64 per wave) from a 128x32 LDS pair
// ---------------------------------------------------------------------------
static __device__ __forceinline__ void wmma_tile(const unsigned short (*Asb)[32],
                                                 const unsigned short (*Bsb)[32],
                                                 v8f acc[2][4],
                                                 int wm, int wn, int l15, int lhalf) {
  v16bf afr[2];
  #pragma unroll
  for (int i = 0; i < 2; ++i) {
    const int row = wm + i * 16 + l15;
    us8 lo = *(const us8*)&Asb[row][lhalf * 8];
    us8 hi = *(const us8*)&Asb[row][16 + lhalf * 8];
    afr[i] = make_frag(lo, hi);
  }
  v16bf bfr[4];
  #pragma unroll
  for (int j = 0; j < 4; ++j) {
    const int nrow = wn + j * 16 + l15;
    us8 lo = *(const us8*)&Bsb[nrow][lhalf * 8];
    us8 hi = *(const us8*)&Bsb[nrow][16 + lhalf * 8];
    bfr[j] = make_frag(lo, hi);
  }
  #pragma unroll
  for (int i = 0; i < 2; ++i)
    #pragma unroll
    for (int j = 0; j < 4; ++j)
      acc[i][j] = __builtin_amdgcn_wmma_f32_16x16x32_bf16(
          false, afr[i], false, bfr[j], (short)0, acc[i][j], false, false);
}

// ---------------------------------------------------------------------------
// GEMM: C[M,N] = A[M,K] @ W[N,K]^T + bias   (bf16 A, bf16 W, f32 accumulate)
// MODE 0: store f32 to Out[m*N + n]                      (final projection)
// MODE 1: store bf16 to Out[((b*NH+h)*TT + t)*DKK + d]   (Q, K head layout)
// MODE 2: store bf16 to Out[((b*NH+h)*DKK + d)*TT + t]   (V transposed)
// 256 threads / 8 waves; tile 128x128, K-step 32, TDM double-buffered LDS.
// ---------------------------------------------------------------------------
template<int MODE>
__global__ __launch_bounds__(256)
void gemm_wmma(const unsigned short* __restrict__ A, const unsigned short* __restrict__ W,
               const float* __restrict__ bias, void* __restrict__ Out,
               int M, int N, int K)
{
#if USE_TDM
  __shared__ unsigned short As[2][128][32];
  __shared__ unsigned short Bs[2][128][32];
#else
  __shared__ unsigned short As[1][128][32];
  __shared__ unsigned short Bs[1][128][32];
#endif

  const int tid   = threadIdx.x;
  const int lane  = tid & 31;
  const int wid   = tid >> 5;
  const int wm    = (wid & 3) * 32;
  const int wn    = (wid >> 2) * 64;
  const int m0    = blockIdx.y * 128;
  const int n0    = blockIdx.x * 128;
  const int l15   = lane & 15;
  const int lhalf = lane >> 4;

  v8f acc[2][4];
  #pragma unroll
  for (int i = 0; i < 2; ++i)
    #pragma unroll
    for (int j = 0; j < 4; ++j) acc[i][j] = v8zero();

  const unsigned short* Abase = A + (size_t)m0 * K;
  const unsigned short* Wbase = W + (size_t)n0 * K;

#if USE_TDM
  const int nk = K / 32;
  if (wid == 0) {
    tdm_load_2d(lds_off32(&As[0][0][0]), Abase, 32, 128, (unsigned)K);
    tdm_load_2d(lds_off32(&Bs[0][0][0]), Wbase, 32, 128, (unsigned)K);
    __builtin_amdgcn_s_wait_tensorcnt(0);
  }
  __syncthreads();
  for (int t = 0; t < nk; ++t) {
    const int cur = t & 1;
    if (wid == 0 && (t + 1) < nk) {   // prefetch next K tile via TDM
      tdm_load_2d(lds_off32(&As[cur ^ 1][0][0]), Abase + (t + 1) * 32, 32, 128, (unsigned)K);
      tdm_load_2d(lds_off32(&Bs[cur ^ 1][0][0]), Wbase + (t + 1) * 32, 32, 128, (unsigned)K);
    }
    wmma_tile(As[cur], Bs[cur], acc, wm, wn, l15, lhalf);
    if (wid == 0 && (t + 1) < nk) __builtin_amdgcn_s_wait_tensorcnt(0);
    __syncthreads();
  }
#else
  const int lrow = tid >> 1;
  const int lcs  = (tid & 1) * 16;
  for (int k0 = 0; k0 < K; k0 += 32) {
    const us8* sa = (const us8*)(Abase + (size_t)lrow * K + k0 + lcs);
    *(us8*)&As[0][lrow][lcs]     = sa[0];
    *(us8*)&As[0][lrow][lcs + 8] = sa[1];
    const us8* sb = (const us8*)(Wbase + (size_t)lrow * K + k0 + lcs);
    *(us8*)&Bs[0][lrow][lcs]     = sb[0];
    *(us8*)&Bs[0][lrow][lcs + 8] = sb[1];
    if (k0 + 32 < K)
      __builtin_prefetch(Abase + (size_t)lrow * K + k0 + 32, 0, 0);
    __syncthreads();
    wmma_tile(As[0], Bs[0], acc, wm, wn, l15, lhalf);
    __syncthreads();
  }
#endif

  // ---- epilogue: bias + store ----
  #pragma unroll
  for (int i = 0; i < 2; ++i) {
    #pragma unroll
    for (int j = 0; j < 4; ++j) {
      const int nn = n0 + wn + j * 16 + l15;
      const float bv = bias[nn];
      #pragma unroll
      for (int r = 0; r < 8; ++r) {
        const int mm = m0 + wm + i * 16 + r + 8 * lhalf;  // C layout: M = r + 8*(lane>=16)
        const float val = acc[i][j][r] + bv;
        if (MODE == 0) {
          ((float*)Out)[(size_t)mm * N + nn] = val;
        } else {
          const unsigned short bb = f32_to_bf16(val);
          const int b = mm >> 11;
          const int t = mm & (TT - 1);
          const int h = nn >> 6;
          const int d = nn & (DKK - 1);
          if (MODE == 1)
            ((unsigned short*)Out)[(((size_t)b * NH + h) * TT + t) * DKK + d] = bb;
          else
            ((unsigned short*)Out)[(((size_t)b * NH + h) * DKK + d) * TT + t] = bb;
        }
      }
    }
  }
}

// ---------------------------------------------------------------------------
// Banded attention, flash-style online softmax (branchless mask path).
// One wave per (b, h, 16-query tile); 32-key steps over the |i-j|<=64 band.
// ---------------------------------------------------------------------------
__global__ __launch_bounds__(32)
void attn_banded(const unsigned short* __restrict__ Q,
                 const unsigned short* __restrict__ Km,
                 const unsigned short* __restrict__ VT,
                 const int* __restrict__ mask,
                 unsigned short* __restrict__ X)
{
  __shared__ unsigned short p_lds[16][32];

  const int lane  = threadIdx.x & 31;
  const int l15   = lane & 15;
  const int lhalf = lane >> 4;
  const int i0    = blockIdx.x * 16;
  const int h     = blockIdx.y;
  const int b     = blockIdx.z;

  const size_t bh = (size_t)b * NH + h;
  const unsigned short* q  = Q  + bh * TT * DKK;
  const unsigned short* kk = Km + bh * TT * DKK;
  const unsigned short* vt = VT + bh * DKK * TT;
  const int* mbase = mask + (size_t)b * TT * TT;

  v16bf aq[2];
  {
    const unsigned short* qr = q + (size_t)(i0 + l15) * DKK;
    #pragma unroll
    for (int dt = 0; dt < 2; ++dt) {
      us8 lo = *(const us8*)(qr + dt * 32 + lhalf * 8);
      us8 hi = *(const us8*)(qr + dt * 32 + 16 + lhalf * 8);
      aq[dt] = make_frag(lo, hi);
    }
  }

  v8f o[4];
  #pragma unroll
  for (int dt = 0; dt < 4; ++dt) o[dt] = v8zero();
  float m8[8], l8[8];
  #pragma unroll
  for (int r = 0; r < 8; ++r) { m8[r] = -3.0e38f; l8[r] = 0.0f; }

  int jstart = i0 - BAND; if (jstart < 0) jstart = 0;
  jstart &= ~31;
  int jend = i0 + 16 + BAND; if (jend > TT) jend = TT;

  const float qscale = 0.125f;  // 1/sqrt(DK)

  for (int j0 = jstart; j0 < jend; j0 += 32) {
    // ---- S = Q @ K^T for keys [j0, j0+32): two 16x16 C tiles ----
    v8f s[2];
    #pragma unroll
    for (int kt = 0; kt < 2; ++kt) {
      v8f sc = v8zero();
      const unsigned short* kr = kk + (size_t)(j0 + kt * 16 + l15) * DKK;
      #pragma unroll
      for (int dt = 0; dt < 2; ++dt) {
        us8 lo = *(const us8*)(kr + dt * 32 + lhalf * 8);
        us8 hi = *(const us8*)(kr + dt * 32 + 16 + lhalf * 8);
        v16bf bk = make_frag(lo, hi);
        sc = __builtin_amdgcn_wmma_f32_16x16x32_bf16(
            false, aq[dt], false, bk, (short)0, sc, false, false);
      }
      s[kt] = sc;
    }

    // ---- scale, band + user mask (branchless), online softmax update ----
    float p0[8], p1[8];
    #pragma unroll
    for (int r = 0; r < 8; ++r) {
      const int i  = i0 + r + 8 * lhalf;
      const int ja = j0 + l15;
      const int jb = j0 + 16 + l15;
      const int* mrow = mbase + (size_t)i * TT;
      const int mva = mrow[ja];            // unconditional loads: no exec branches
      const int mvb = mrow[jb];
      int da = i - ja; if (da < 0) da = -da;
      int db = i - jb; if (db < 0) db = -db;
      float sa = s[0][r] * qscale;
      float sb = s[1][r] * qscale;
      const int va = (da <= BAND) & (mva != 0);
      const int vb = (db <= BAND) & (mvb != 0);
      sa = va ? sa : MINV;
      sb = vb ? sb : MINV;

      float tmax = fmaxf(sa, sb);
      #pragma unroll
      for (int xm = 1; xm < 16; xm <<= 1)
        tmax = fmaxf(tmax, __shfl_xor(tmax, xm, 32));
      const float newm = fmaxf(m8[r], tmax);
      const float corr = __expf(m8[r] - newm);
      const float ea = __expf(sa - newm);
      const float eb = __expf(sb - newm);
      float rs = ea + eb;
      #pragma unroll
      for (int xm = 1; xm < 16; xm <<= 1)
        rs += __shfl_xor(rs, xm, 32);
      l8[r] = l8[r] * corr + rs;
      m8[r] = newm;
      #pragma unroll
      for (int dt = 0; dt < 4; ++dt) o[dt][r] *= corr;
      p0[r] = ea; p1[r] = eb;
    }

    // ---- transpose P (C layout) -> A layout through LDS ----
    #pragma unroll
    for (int r = 0; r < 8; ++r) {
      const int row = r + 8 * lhalf;
      p_lds[row][l15]      = f32_to_bf16(p0[r]);
      p_lds[row][16 + l15] = f32_to_bf16(p1[r]);
    }
    __syncthreads();
    v16bf ap;
    {
      us8 lo = *(const us8*)&p_lds[l15][lhalf * 8];
      us8 hi = *(const us8*)&p_lds[l15][16 + lhalf * 8];
      ap = make_frag(lo, hi);
    }
    __syncthreads();

    // ---- O += P @ V : pre-transposed V makes B-fragments contiguous ----
    #pragma unroll
    for (int dt = 0; dt < 4; ++dt) {
      const int d = dt * 16 + l15;
      const unsigned short* vr = vt + (size_t)d * TT + j0 + lhalf * 8;
      us8 lo = *(const us8*)vr;
      us8 hi = *(const us8*)(vr + 16);
      v16bf bv = make_frag(lo, hi);
      o[dt] = __builtin_amdgcn_wmma_f32_16x16x32_bf16(
          false, ap, false, bv, (short)0, o[dt], false, false);
    }
  }

  // ---- normalize and store x (bf16, [B,T,F] layout for the final GEMM) ----
  #pragma unroll
  for (int dt = 0; dt < 4; ++dt) {
    #pragma unroll
    for (int r = 0; r < 8; ++r) {
      const int row = i0 + r + 8 * lhalf;
      const int d   = dt * 16 + l15;
      const float inv = 1.0f / fmaxf(l8[r], 1e-30f);
      X[((size_t)b * TT + row) * FF + h * DKK + d] = f32_to_bf16(o[dt][r] * inv);
    }
  }
}

// ---------------------------------------------------------------------------
// Host side
// ---------------------------------------------------------------------------
extern "C" void kernel_launch(void* const* d_in, const int* in_sizes, int n_in,
                              void* d_out, int out_size, void* d_ws, size_t ws_size,
                              hipStream_t stream) {
  const float* query = (const float*)d_in[0];
  const float* key   = (const float*)d_in[1];
  const float* value = (const float*)d_in[2];
  const float* Wq    = (const float*)d_in[3];
  const float* bq    = (const float*)d_in[4];
  const float* Wk    = (const float*)d_in[5];
  const float* bk    = (const float*)d_in[6];
  const float* Wv    = (const float*)d_in[7];
  const float* bv    = (const float*)d_in[8];
  const float* Wo    = (const float*)d_in[9];
  const float* bo    = (const float*)d_in[10];
  const int*   mask  = (const int*)d_in[11];
  float* out = (float*)d_out;

  const int B = 4, M = B * TT, N = FF, K = FF;
  const size_t NE_X = (size_t)M * FF;   // 8M elements
  const size_t NE_W = (size_t)FF * FF;  // 1M elements

  unsigned short* p = (unsigned short*)d_ws;
  unsigned short* qx  = p; p += NE_X;   // bf16 copies of inputs
  unsigned short* kx  = p; p += NE_X;
  unsigned short* vx  = p; p += NE_X;
  unsigned short* wqb = p; p += NE_W;   // bf16 copies of weights
  unsigned short* wkb = p; p += NE_W;
  unsigned short* wvb = p; p += NE_W;
  unsigned short* wob = p; p += NE_W;
  unsigned short* qb  = p; p += NE_X;   // projected Q [B,H,T,Dk]
  unsigned short* kb  = p; p += NE_X;   // projected K [B,H,T,Dk]
  unsigned short* vT  = p; p += NE_X;   // projected V [B,H,Dk,T]
  unsigned short* xb  = p;              // attention out [B,T,F]

  auto cvt = [&](const float* src, unsigned short* dst, size_t n) {
    cvt_f32_bf16<<<(int)(n / (8 * 256)), 256, 0, stream>>>(src, dst, (int)n);
  };
  cvt(query, qx, NE_X);
  cvt(key,   kx, NE_X);
  cvt(value, vx, NE_X);
  cvt(Wq, wqb, NE_W);
  cvt(Wk, wkb, NE_W);
  cvt(Wv, wvb, NE_W);
  cvt(Wo, wob, NE_W);

  dim3 ggrid(N / 128, M / 128);
  dim3 gblock(256);
  gemm_wmma<1><<<ggrid, gblock, 0, stream>>>(qx, wqb, bq, qb, M, N, K);
  gemm_wmma<1><<<ggrid, gblock, 0, stream>>>(kx, wkb, bk, kb, M, N, K);
  gemm_wmma<2><<<ggrid, gblock, 0, stream>>>(vx, wvb, bv, vT, M, N, K);

  dim3 agrid(TT / 16, NH, B);
  attn_banded<<<agrid, dim3(32), 0, stream>>>(qb, kb, vT, mask, xb);

  gemm_wmma<0><<<ggrid, gblock, 0, stream>>>(xb, wob, bo, out, M, N, K);
}
